// CogVLMDecoderLayer_2886218023370
// MI455X (gfx1250) — matrix operand order repaired and checked by reading, not compiled, not run on hardware
//
#include <hip/hip_runtime.h>

// ---------------------------------------------------------------- constants
static constexpr int S   = 2048;
static constexpr int H   = 4096;
static constexpr int NH  = 32;
static constexpr int HD  = 128;
static constexpr int II  = 11008;
static constexpr int H3  = 3 * H;      // 12288
static constexpr int I2  = 2 * II;     // 22016
static constexpr int SPLIT = 1024;     // rows [0,1024) vision, [1024,2048) lang

typedef __bf16  bf16x8  __attribute__((ext_vector_type(8)));
typedef __bf16  bf16x16 __attribute__((ext_vector_type(16)));
typedef float   f32x8   __attribute__((ext_vector_type(8)));

__device__ __forceinline__ f32x8 wmma_bf16(bf16x16 a, bf16x16 b, f32x8 c) {
    return __builtin_amdgcn_wmma_f32_16x16x32_bf16(
        /*neg_a=*/false, a, /*neg_b=*/false, b,
        /*c_mod=*/(short)0, c, /*reuse_a=*/false, /*reuse_b=*/false);
}

// contiguous 8x bf16 (16B) -> fragment elements [o, o+8)
__device__ __forceinline__ void load_bf8(bf16x16& f, int o, const __bf16* p) {
    bf16x8 t = *reinterpret_cast<const bf16x8*>(p);
#pragma unroll
    for (int j = 0; j < 8; ++j) f[o + j] = t[j];
}

// 32B (2x b128) async global->LDS copy; IOFFSET applies to both addresses.
__device__ __forceinline__ void async_copy_32B(void* lds, const void* gsrc) {
    const unsigned            l = (unsigned)(size_t)lds;           // low 32 bits used as LDS address
    const unsigned long long  g = (unsigned long long)(size_t)gsrc;
    asm volatile(
        "global_load_async_to_lds_b128 %0, %1, off\n\t"
        "global_load_async_to_lds_b128 %0, %1, off offset:16"
        :: "v"(l), "v"(g) : "memory");
}
__device__ __forceinline__ void wait_async0() {
    asm volatile("s_wait_asynccnt 0x0" ::: "memory");
}

// =====================================================================
// RMSNorm: x = hid (+ add), out_bf = x * rsqrt(mean(x^2)+eps) * w
// optionally stores x (the new residual) as f32.
// block = 256 threads, one block per row, H = 4096 -> 16 elems/thread.
// =====================================================================
__global__ void rmsnorm_kernel(const float* __restrict__ hid,
                               const float* __restrict__ add,   // nullable
                               const float* __restrict__ w,
                               __bf16* __restrict__ out_bf,
                               float* __restrict__ resid_out) { // nullable
    const int row = blockIdx.x;
    const int tid = threadIdx.x;
    const float* xr = hid + (size_t)row * H;
    const float* ar = add ? add + (size_t)row * H : nullptr;

    float v[16];
    float ss = 0.f;
#pragma unroll
    for (int i = 0; i < 16; ++i) {
        int idx = i * 256 + tid;
        float t = xr[idx];
        if (ar) t += ar[idx];
        v[i] = t;
        ss += t * t;
    }
#pragma unroll
    for (int off = 16; off >= 1; off >>= 1) ss += __shfl_xor(ss, off);
    __shared__ float red[8];
    if ((tid & 31) == 0) red[tid >> 5] = ss;
    __syncthreads();
    float tot = 0.f;
#pragma unroll
    for (int i = 0; i < 8; ++i) tot += red[i];
    const float scale = rsqrtf(tot * (1.0f / (float)H) + 1e-5f);

#pragma unroll
    for (int i = 0; i < 16; ++i) {
        int idx = i * 256 + tid;
        out_bf[(size_t)row * H + idx] = (__bf16)(v[i] * scale * w[idx]);
        if (resid_out) resid_out[(size_t)row * H + idx] = v[i];
    }
}

// =====================================================================
// GEMM: C[M,N] (f32) = A[M,K] (bf16 row-major) @ B[K,N] (f32 weights,
// converted to bf16 on the fly). Expert select per 128-row tile.
//
// Grid mapping: blockIdx.x = M-tile (16), blockIdx.y = N-tile.
// Consecutive blocks share one B column-panel (~2 MB) -> each weight
// byte is fetched from HBM once; re-reads hit the 192 MB L2.
//
// Block: 256 thr (8 waves), tile 128x128x32, double-buffered LDS.
// A tile streamed with GLOBAL_LOAD_ASYNC_TO_LDS_B128 (ASYNCcnt);
// B tile via regs (f32->bf16 convert) stored transposed in LDS.
// Wave grid 4(M) x 2(N): 2 M-frags x 4 N-frags of 16x16 per wave.
// =====================================================================
__global__ void gemm_bf16_kernel(const __bf16* __restrict__ A,
                                 const float* __restrict__ Bv,
                                 const float* __restrict__ Bl,
                                 float* __restrict__ C,
                                 int K, int N) {
    constexpr int BM = 128, BN = 128, BK = 32;
    const int m0 = blockIdx.x * BM;
    const int n0 = blockIdx.y * BN;
    const float* __restrict__ Bg = (m0 < SPLIT) ? Bv : Bl;

    const int tid  = threadIdx.x;
    const int wave = tid >> 5;
    const int lane = tid & 31;
    const int ln   = lane & 15;
    const int lh   = lane >> 4;
    const int waveM = wave & 3;   // 0..3  -> 32 rows each
    const int waveN = wave >> 2;  // 0..1  -> 64 cols each

    __shared__ __align__(16) __bf16 As[2][BM][BK];   // 2 x 8 KB
    __shared__ __align__(16) __bf16 Bs[2][BN][BK];   // 2 x 8 KB (transposed)

    f32x8 acc[2][4];
#pragma unroll
    for (int i = 0; i < 2; ++i)
#pragma unroll
        for (int j = 0; j < 4; ++j)
#pragma unroll
            for (int r = 0; r < 8; ++r) acc[i][j][r] = 0.f;

    // per-thread load coordinates
    const int arow = tid >> 1;               // 0..127
    const int acol = (tid & 1) * 16;         // 0 or 16
    const int brow = tid >> 3;               // 0..31   (k within tile)
    const int bcol = (tid & 7) * 16;         // 0..112  (n within tile)

    auto issueA = [&](int t, int buf) {
        async_copy_32B(&As[buf][arow][acol],
                       A + (size_t)(m0 + arow) * K + t * BK + acol);
    };
    auto loadB = [&](int t, float4* r) {
        const float* src = Bg + (size_t)(t * BK + brow) * N + n0 + bcol;
#pragma unroll
        for (int q = 0; q < 4; ++q) r[q] = *reinterpret_cast<const float4*>(src + 4 * q);
    };
    auto storeB = [&](int buf, const float4* r) {
#pragma unroll
        for (int q = 0; q < 4; ++q) {
            Bs[buf][bcol + 4 * q + 0][brow] = (__bf16)r[q].x;
            Bs[buf][bcol + 4 * q + 1][brow] = (__bf16)r[q].y;
            Bs[buf][bcol + 4 * q + 2][brow] = (__bf16)r[q].z;
            Bs[buf][bcol + 4 * q + 3][brow] = (__bf16)r[q].w;
        }
    };

    const int T = K / BK;

    // ---- prologue: tile 0 into buffer 0
    {
        issueA(0, 0);
        float4 b0[4];
        loadB(0, b0);
        storeB(0, b0);
    }
    wait_async0();
    __syncthreads();

    for (int t = 0; t < T; ++t) {
        const int buf = t & 1;
        const bool more = (t + 1 < T);

        // ---- issue next tile (overlaps with this tile's WMMAs)
        float4 breg[4];
        if (more) {
            issueA(t + 1, 1 - buf);
            loadB(t + 1, breg);
            if (t + 2 < T)
                __builtin_prefetch(Bg + (size_t)((t + 2) * BK + brow) * N + n0 + bcol, 0, 1);
        }

        // ---- compute on buffer `buf`
        bf16x16 af[2], bfr[4];
        const int base0 = lh * 8;
#pragma unroll
        for (int i = 0; i < 2; ++i) {
            const int r = waveM * 32 + i * 16 + ln;
            load_bf8(af[i], 0, &As[buf][r][base0]);
            load_bf8(af[i], 8, &As[buf][r][16 + base0]);
        }
        const int kh = lh * 8;
#pragma unroll
        for (int j = 0; j < 4; ++j) {
            const int c = waveN * 64 + j * 16 + ln;
            load_bf8(bfr[j], 0, &Bs[buf][c][kh]);
            load_bf8(bfr[j], 8, &Bs[buf][c][kh + 16]);
        }
#pragma unroll
        for (int i = 0; i < 2; ++i)
#pragma unroll
            for (int j = 0; j < 4; ++j)
                acc[i][j] = wmma_bf16(af[i], bfr[j], acc[i][j]);

        // ---- publish next B tile, close the pipeline stage
        if (more) storeB(1 - buf, breg);
        wait_async0();
        __syncthreads();
    }

    // ---- epilogue: C layout row = r + 8*lh, col = ln
#pragma unroll
    for (int i = 0; i < 2; ++i)
#pragma unroll
        for (int j = 0; j < 4; ++j)
#pragma unroll
            for (int r = 0; r < 8; ++r) {
                const int gm = m0 + waveM * 32 + i * 16 + r + 8 * lh;
                const int gn = n0 + waveN * 64 + j * 16 + ln;
                C[(size_t)gm * N + gn] = acc[i][j][r];
            }
}

// =====================================================================
// RoPE + split + V transpose.
// qkv (S, 3H) f32.  Writes q_bf,k_bf (S,H) bf16 and vt (NH*HD, S) bf16.
// One thread per (s, h, d2<64) handling the rotate_half pair.
// =====================================================================
__global__ void rope_kernel(const float* __restrict__ qkv,
                            const float* __restrict__ cosb,
                            const float* __restrict__ sinb,
                            __bf16* __restrict__ qb,
                            __bf16* __restrict__ kb,
                            __bf16* __restrict__ vt) {
    const int gid = blockIdx.x * blockDim.x + threadIdx.x;
    if (gid >= S * NH * 64) return;
    const int d2 = gid & 63;
    const int h  = (gid >> 6) & 31;
    const int s  = gid >> 11;

    const size_t base = (size_t)s * H3 + h * HD;
    const float q1 = qkv[base + d2],          q2 = qkv[base + 64 + d2];
    const float k1 = qkv[base + H + d2],      k2 = qkv[base + H + 64 + d2];
    const float v1 = qkv[base + 2 * H + d2],  v2 = qkv[base + 2 * H + 64 + d2];

    const float c1 = cosb[(size_t)s * HD + d2],      s1 = sinb[(size_t)s * HD + d2];
    const float c2 = cosb[(size_t)s * HD + 64 + d2], s2 = sinb[(size_t)s * HD + 64 + d2];

    const size_t o = (size_t)s * H + h * HD;
    qb[o + d2]      = (__bf16)(q1 * c1 - q2 * s1);
    qb[o + 64 + d2] = (__bf16)(q2 * c2 + q1 * s2);
    kb[o + d2]      = (__bf16)(k1 * c1 - k2 * s1);
    kb[o + 64 + d2] = (__bf16)(k2 * c2 + k1 * s2);

    vt[(size_t)(h * HD + d2) * S + s]      = (__bf16)v1;
    vt[(size_t)(h * HD + 64 + d2) * S + s] = (__bf16)v2;
}

// =====================================================================
// Flash attention, causal. Grid (S/64, NH), block 128 (4 waves).
// Each wave owns 16 q-rows, loops 32-wide key tiles, online softmax.
// QK^T and P.V via WMMA; V pre-transposed so PV B-frag is contiguous.
// =====================================================================
__global__ void flash_attn_kernel(const __bf16* __restrict__ Q,
                                  const __bf16* __restrict__ Kp,
                                  const __bf16* __restrict__ Vt,
                                  __bf16* __restrict__ O) {
    constexpr float SCALE = 0.08838834764831845f;  // 1/sqrt(128)
    constexpr float NEG   = -1e30f;

    const int wave = threadIdx.x >> 5;
    const int lane = threadIdx.x & 31;
    const int ln = lane & 15, lh = lane >> 4;
    const int h  = blockIdx.y;
    const int q0 = blockIdx.x * 64 + wave * 16;

    __shared__ __align__(16) __bf16 Pbuf[4][16 * 32];  // per-wave P staging

    // Q fragments: 16 rows x 128 d, 4 k-steps of 32
    bf16x16 qf[4];
    {
        const __bf16* qrow = Q + (size_t)(q0 + ln) * H + h * HD;
        const int base0 = lh * 8;
#pragma unroll
        for (int kk = 0; kk < 4; ++kk) {
            load_bf8(qf[kk], 0, qrow + kk * 32 + base0);
            load_bf8(qf[kk], 8, qrow + kk * 32 + 16 + base0);
        }
    }

    f32x8 acc[8];
#pragma unroll
    for (int t = 0; t < 8; ++t)
#pragma unroll
        for (int r = 0; r < 8; ++r) acc[t][r] = 0.f;
    float mrow[8], lrow[8];
#pragma unroll
    for (int r = 0; r < 8; ++r) { mrow[r] = NEG; lrow[r] = 0.f; }

    const int nkt = (q0 + 15) / 32 + 1;
    for (int kt = 0; kt < nkt; ++kt) {
        const int kbase = kt * 32;

        // ---- scores: S = Q K^T  (16 x 32)
        f32x8 sc[2];
#pragma unroll
        for (int c = 0; c < 2; ++c)
#pragma unroll
            for (int r = 0; r < 8; ++r) sc[c][r] = 0.f;

#pragma unroll
        for (int c = 0; c < 2; ++c) {
            const int key = kbase + c * 16 + ln;
            const __bf16* krow = Kp + (size_t)key * H + h * HD;
            const int kh = lh * 8;
#pragma unroll
            for (int kk = 0; kk < 4; ++kk) {
                bf16x16 bfrag;
                load_bf8(bfrag, 0, krow + kk * 32 + kh);
                load_bf8(bfrag, 8, krow + kk * 32 + kh + 16);
                sc[c] = wmma_bf16(qf[kk], bfrag, sc[c]);
            }
        }

        // ---- online softmax (row stats shared across 16-lane halves)
        float alpha[8];
#pragma unroll
        for (int r = 0; r < 8; ++r) {
            const int qidx = q0 + r + 8 * lh;
            float s0 = sc[0][r] * SCALE;
            float s1 = sc[1][r] * SCALE;
            if (kbase + ln > qidx)      s0 = NEG;
            if (kbase + 16 + ln > qidx) s1 = NEG;
            float mx = fmaxf(s0, s1);
#pragma unroll
            for (int off = 1; off < 16; off <<= 1) mx = fmaxf(mx, __shfl_xor(mx, off));
            const float mnew = fmaxf(mrow[r], mx);
            alpha[r] = __expf(mrow[r] - mnew);
            const float p0 = __expf(s0 - mnew);
            const float p1 = __expf(s1 - mnew);
            float rs = p0 + p1;
#pragma unroll
            for (int off = 1; off < 16; off <<= 1) rs += __shfl_xor(rs, off);
            lrow[r] = lrow[r] * alpha[r] + rs;
            mrow[r] = mnew;
            sc[0][r] = p0;
            sc[1][r] = p1;
        }

        // ---- P (C layout) -> LDS -> A layout
#pragma unroll
        for (int c = 0; c < 2; ++c)
#pragma unroll
            for (int r = 0; r < 8; ++r) {
                const int m = r + 8 * lh;
                Pbuf[wave][m * 32 + c * 16 + ln] = (__bf16)sc[c][r];
            }
        __builtin_amdgcn_wave_barrier();

        bf16x16 pf;
        {
            const int base0 = lh * 8;
            load_bf8(pf, 0, &Pbuf[wave][ln * 32 + base0]);
            load_bf8(pf, 8, &Pbuf[wave][ln * 32 + 16 + base0]);
        }
        __builtin_amdgcn_wave_barrier();

        // ---- acc = alpha*acc + P @ V
#pragma unroll
        for (int t = 0; t < 8; ++t) {
#pragma unroll
            for (int r = 0; r < 8; ++r) acc[t][r] *= alpha[r];
            bf16x16 vfrag;
            const __bf16* vrow = Vt + (size_t)(h * HD + t * 16 + ln) * S;
            const int kh = lh * 8;
            load_bf8(vfrag, 0, vrow + kbase + kh);
            load_bf8(vfrag, 8, vrow + kbase + kh + 16);
            acc[t] = wmma_bf16(pf, vfrag, acc[t]);
        }
    }

    // ---- normalize + store (bf16, layout (S,H))
#pragma unroll
    for (int t = 0; t < 8; ++t)
#pragma unroll
        for (int r = 0; r < 8; ++r) {
            const int gm = q0 + r + 8 * lh;
            const int gc = h * HD + t * 16 + ln;
            O[(size_t)gm * H + gc] = (__bf16)(acc[t][r] / lrow[r]);
        }
}

// =====================================================================
// SwiGLU gate: h = silu(g) * u, gu is (S, 2I) f32, h is (S, I) bf16
// =====================================================================
__global__ void silu_mul_kernel(const float* __restrict__ gu,
                                __bf16* __restrict__ hb) {
    const int gid = blockIdx.x * blockDim.x + threadIdx.x;
    if (gid >= S * II) return;
    const int s = gid / II;
    const int c = gid - s * II;
    const float g = gu[(size_t)s * I2 + c];
    const float u = gu[(size_t)s * I2 + II + c];
    const float sg = g / (1.0f + __expf(-g));
    hb[gid] = (__bf16)(sg * u);
}

// =====================================================================
extern "C" void kernel_launch(void* const* d_in, const int* in_sizes, int n_in,
                              void* d_out, int out_size, void* d_ws, size_t ws_size,
                              hipStream_t stream) {
    (void)in_sizes; (void)n_in; (void)out_size; (void)ws_size;
    const float* hidden = (const float*)d_in[0];
    // d_in[1], d_in[2]: vision_ids/lang_ids -> contiguous halves, handled by row split
    const float* w_ln1  = (const float*)d_in[3];
    const float* w_ln2  = (const float*)d_in[4];
    const float* wqkv_l = (const float*)d_in[5];
    const float* wqkv_v = (const float*)d_in[6];
    const float* wo_l   = (const float*)d_in[7];
    const float* wo_v   = (const float*)d_in[8];
    const float* wgu_l  = (const float*)d_in[9];
    const float* wgu_v  = (const float*)d_in[10];
    const float* wd_l   = (const float*)d_in[11];
    const float* wd_v   = (const float*)d_in[12];
    const float* cosb   = (const float*)d_in[13];
    const float* sinb   = (const float*)d_in[14];

    float* out_f   = (float*)d_out;                       // (S,H) mlp output
    float* resid_f = (float*)d_out + (size_t)S * H;       // (S,H) residual

    char* ws = (char*)d_ws;
    size_t off = 0;
    auto alloc = [&](size_t bytes) { char* p = ws + off; off = (off + bytes + 255) & ~(size_t)255; return p; };
    __bf16* x1    = (__bf16*)alloc((size_t)S * H * 2);
    float*  qkv   = (float*) alloc((size_t)S * H3 * 4);
    __bf16* qb    = (__bf16*)alloc((size_t)S * H * 2);
    __bf16* kb    = (__bf16*)alloc((size_t)S * H * 2);
    __bf16* vt    = (__bf16*)alloc((size_t)S * H * 2);
    __bf16* attnb = (__bf16*)alloc((size_t)S * H * 2);
    float*  oproj = (float*) alloc((size_t)S * H * 4);
    __bf16* x2    = (__bf16*)alloc((size_t)S * H * 2);
    float*  gu    = (float*) alloc((size_t)S * I2 * 4);
    __bf16* hb    = (__bf16*)alloc((size_t)S * II * 2);

    // 1) x1 = rmsnorm(hidden, w_ln1)
    rmsnorm_kernel<<<S, 256, 0, stream>>>(hidden, nullptr, w_ln1, x1, nullptr);
    // 2) qkv = x1 @ wqkv_{v|l}   (M=2048, K=4096, N=12288)
    gemm_bf16_kernel<<<dim3(S / 128, H3 / 128), 256, 0, stream>>>(x1, wqkv_v, wqkv_l, qkv, H, H3);
    // 3) RoPE + split + V transpose
    rope_kernel<<<(S * NH * 64 + 255) / 256, 256, 0, stream>>>(qkv, cosb, sinb, qb, kb, vt);
    // 4) causal flash attention
    flash_attn_kernel<<<dim3(S / 64, NH), 128, 0, stream>>>(qb, kb, vt, attnb);
    // 5) oproj = attn @ wo_{v|l}
    gemm_bf16_kernel<<<dim3(S / 128, H / 128), 256, 0, stream>>>(attnb, wo_v, wo_l, oproj, H, H);
    // 6) resid = hidden + oproj ; x2 = rmsnorm(resid, w_ln2)
    rmsnorm_kernel<<<S, 256, 0, stream>>>(hidden, oproj, w_ln2, x2, resid_f);
    // 7) gu = x2 @ wgu_{v|l}   (N=22016)
    gemm_bf16_kernel<<<dim3(S / 128, I2 / 128), 256, 0, stream>>>(x2, wgu_v, wgu_l, gu, H, I2);
    // 8) hb = silu(g) * u
    silu_mul_kernel<<<(S * II + 255) / 256, 256, 0, stream>>>(gu, hb);
    // 9) out = hb @ wd_{v|l}   (K=11008, N=4096)
    gemm_bf16_kernel<<<dim3(S / 128, H / 128), 256, 0, stream>>>(hb, wd_v, wd_l, out_f, II, H);
}
